// VSDANet_60043642798804
// MI455X (gfx1250) — matrix operand order repaired and checked by reading, compiled, not run on hardware
//
#include <hip/hip_runtime.h>
#include <hip/hip_bf16.h>
#include <cstdint>

// ---------------------------------------------------------------------------
// Types for CDNA5 WMMA (wave32, 16x16x32 bf16 -> f32)
// ---------------------------------------------------------------------------
typedef __bf16 bf16_t;
typedef bf16_t v16bf __attribute__((ext_vector_type(16)));
typedef float  v8f   __attribute__((ext_vector_type(8)));

union BfFrag { v16bf v; uint32_t u[8]; };

__device__ inline v8f wmma_bf16(v16bf a, v16bf b, v8f c) {
    // 8 args: (neg_a, A, neg_b, B, c_mod, C, reuse_a, reuse_b)
    return __builtin_amdgcn_wmma_f32_16x16x32_bf16(false, a, false, b,
                                                   (short)0, c, false, false);
}

// Problem constants (from reference)
constexpr int Bb = 8, Nn = 1024, Dd = 256, Hh = 8, DH = 32, H4 = 1024;
constexpr int Mm = Bb * Nn;   // 8192 token rows

// ---------------------------------------------------------------------------
// Tensor Data Mover staging (gfx1250): one wave issues a 2D descriptor that
// DMA-copies a [rows x 32] bf16 tile (row stride = strideElems) into LDS,
// inserting 1 pad dword every 16 dwords (=> 17-dword LDS row stride, bank-
// conflict-free fragment reads).  Tracked with TENSORcnt.
// ---------------------------------------------------------------------------
#define LDSR 17                          // padded LDS row stride in dwords
#define ABROWS 64
#define BROWS  128
#define AB_TILE (ABROWS * LDSR)          // B region starts here (dwords)
#define BUFSZ ((ABROWS + BROWS) * LDSR)  // one double-buffer half (dwords)

#if defined(__gfx1250__) && __has_builtin(__builtin_amdgcn_tensor_load_to_lds)
#define HAVE_TDM 1
typedef unsigned int u32x4 __attribute__((ext_vector_type(4)));
typedef int          i32x4 __attribute__((ext_vector_type(4)));
typedef int          i32x8 __attribute__((ext_vector_type(8)));

__device__ inline void tdm_stage_2d(uint32_t ldsOffBytes, const bf16_t* gptr,
                                    uint32_t strideElems, uint32_t rows) {
    uint64_t ga = (uint64_t)(uintptr_t)gptr;
    u32x4 g0;
    g0[0] = 1u;                                     // count=1, user descriptor
    g0[1] = ldsOffBytes;                            // lds_addr
    g0[2] = (uint32_t)ga;                           // global_addr[31:0]
    g0[3] = (uint32_t)((ga >> 32) & 0x01FFFFFFu) | (2u << 30); // addr[56:32], type=2
    i32x8 g1;
    // data_size=1 (2B) | pad_enable | pad_interval=3 (16 dwords) | pad_amount=0 (1 dword)
    g1[0] = (int)((1u << 16) | (1u << 20) | (3u << 22));
    g1[1] = (int)((strideElems & 0xFFFFu) << 16);              // tensor_dim0 lo
    g1[2] = (int)(((strideElems >> 16) & 0xFFFFu) |            // tensor_dim0 hi
                  ((rows & 0xFFFFu) << 16));                   // tensor_dim1 lo
    g1[3] = (int)(((rows >> 16) & 0xFFFFu) | (32u << 16));     // dim1 hi, tile_dim0=32
    g1[4] = (int)(rows & 0xFFFFu);                             // tile_dim1, tile_dim2=0
    g1[5] = (int)strideElems;                                  // tensor_dim0_stride lo32
    g1[6] = 0;                                                 // stride hi bits
    g1[7] = 0;
    i32x4 z4 = {0, 0, 0, 0};
#if __clang_major__ >= 23
    i32x8 z8 = {0, 0, 0, 0, 0, 0, 0, 0};
    __builtin_amdgcn_tensor_load_to_lds(g0, g1, z4, z4, z8, 0);
#else
    __builtin_amdgcn_tensor_load_to_lds(g0, g1, z4, z4, 0);
#endif
}
#endif

// ---------------------------------------------------------------------------
// LayerNorm (fp32 in) -> bf16 out.  One block (256 thr) per row, D = 256.
// ---------------------------------------------------------------------------
__global__ __launch_bounds__(256)
void k_ln(const float* __restrict__ x, const float* __restrict__ g,
          const float* __restrict__ b, bf16_t* __restrict__ out) {
    __shared__ float red[256];
    int row = blockIdx.x, tid = threadIdx.x;
    float v = x[(size_t)row * Dd + tid];
    red[tid] = v; __syncthreads();
    for (int s = 128; s > 0; s >>= 1) {
        if (tid < s) red[tid] += red[tid + s];
        __syncthreads();
    }
    float mean = red[0] * (1.0f / Dd);
    __syncthreads();
    float dv = v - mean;
    red[tid] = dv * dv; __syncthreads();
    for (int s = 128; s > 0; s >>= 1) {
        if (tid < s) red[tid] += red[tid + s];
        __syncthreads();
    }
    float var = red[0] * (1.0f / Dd);
    float r = dv * rsqrtf(var + 1e-5f) * g[tid] + b[tid];
    out[(size_t)row * Dd + tid] = (bf16_t)r;
}

// ---------------------------------------------------------------------------
// Weight prep: fp32 [K,Nt] -> bf16 [Nt,K] (transposed, K-contiguous), and
// straight fp32 -> bf16 convert for weights already in [out,in] order.
// ---------------------------------------------------------------------------
__global__ void k_transpose_bf16(const float* __restrict__ W,
                                 bf16_t* __restrict__ WT, int K, int Nt) {
    int idx = blockIdx.x * 256 + threadIdx.x;
    if (idx >= K * Nt) return;
    int n = idx / K, k = idx % K;
    WT[idx] = (bf16_t)W[(size_t)k * Nt + n];
}

__global__ void k_convert_bf16(const float* __restrict__ src,
                               bf16_t* __restrict__ dst, int cnt) {
    int idx = blockIdx.x * 256 + threadIdx.x;
    if (idx < cnt) dst[idx] = (bf16_t)src[idx];
}

// ---------------------------------------------------------------------------
// GEMM: out[M,Nt](f32) = A[M,K](bf16) @ WT[Nt,K](bf16)^T + bias (+ residual)
// Block = 256 thr = 8 waves, tile 64(M) x 128(N); each wave: 32x32 via
// 2x2 v_wmma_f32_16x16x32_bf16 tiles.  A/B K-tiles double-buffered in LDS by
// the Tensor Data Mover (wave 0 issues descriptors for tile k+1 while all
// waves compute tile k; s_wait_tensorcnt synchronizes), with a cooperative-
// copy fallback when the TDM builtin is unavailable.
// ---------------------------------------------------------------------------
__global__ __launch_bounds__(256)
void k_gemm(const bf16_t* __restrict__ A, const bf16_t* __restrict__ WT,
            const float* __restrict__ bias, const float* __restrict__ res,
            float* __restrict__ out, int K, int Nt) {
    __shared__ uint32_t tile[2 * BUFSZ];   // ~26 KB double-buffered staging
    const uint32_t* A32 = (const uint32_t*)A;
    const uint32_t* W32 = (const uint32_t*)WT;
    int tid = threadIdx.x, lane = tid & 31, wave = tid >> 5;
    int half = lane >> 4, l15 = lane & 15;
    int K2 = K >> 1;
    int rowBase0 = blockIdx.x * 64;
    int colBase0 = blockIdx.y * 128;
    int waveM = (wave >> 2) * 32, waveN = (wave & 3) * 32;

    v8f acc[2][2] = {};
    int cur = 0;
#ifdef HAVE_TDM
    if (wave == 0) {   // prologue: DMA first K-tile into buffer 0
        tdm_stage_2d(0u, A + (size_t)rowBase0 * K, (uint32_t)K, 64u);
        tdm_stage_2d(AB_TILE * 4u, WT + (size_t)colBase0 * K, (uint32_t)K, 128u);
    }
#endif
    for (int k0 = 0; k0 < K; k0 += 32) {
#ifdef HAVE_TDM
        if (wave == 0) __builtin_amdgcn_s_wait_tensorcnt(0);  // buf[cur] ready
        __syncthreads();
        if (wave == 0 && k0 + 32 < K) {   // overlap: DMA next tile -> buf[cur^1]
            int nb = cur ^ 1;
            tdm_stage_2d((uint32_t)(nb * BUFSZ) * 4u,
                         A + (size_t)rowBase0 * K + (k0 + 32), (uint32_t)K, 64u);
            tdm_stage_2d((uint32_t)(nb * BUFSZ + AB_TILE) * 4u,
                         WT + (size_t)colBase0 * K + (k0 + 32), (uint32_t)K, 128u);
        }
#else
        uint32_t* buf = &tile[cur * BUFSZ];
        for (int idx = tid; idx < ABROWS * 16; idx += 256) {
            int r = idx >> 4, c = idx & 15;
            buf[r * LDSR + c] = A32[(size_t)(rowBase0 + r) * K2 + (k0 >> 1) + c];
        }
        for (int idx = tid; idx < BROWS * 16; idx += 256) {
            int r = idx >> 4, c = idx & 15;
            buf[AB_TILE + r * LDSR + c] = W32[(size_t)(colBase0 + r) * K2 + (k0 >> 1) + c];
        }
        __syncthreads();
#endif
        if (k0 + 64 < K) {   // L2 prefetch two tiles ahead (global_prefetch_b8)
            __builtin_prefetch(A32 + (size_t)(rowBase0 + (tid & 63)) * K2 + ((k0 + 64) >> 1), 0, 1);
            if (tid < 128)
                __builtin_prefetch(W32 + (size_t)(colBase0 + tid) * K2 + ((k0 + 64) >> 1), 0, 1);
        }
        // The TDM writes LDS behind the compiler's back (descriptor carries a
        // raw offset).  Escape the pointer + memory clobber so the fragment
        // ds_loads below cannot be folded away as reads of unwritten memory.
        {
            uint32_t* tp = &tile[0];
            asm volatile("" : "+r"(tp) : : "memory");
        }

        const uint32_t* bufA = &tile[cur * BUFSZ];
        const uint32_t* bufB = bufA + AB_TILE;
        BfFrag af[2], bf[2];
#pragma unroll
        for (int r = 0; r < 2; r++) {
            const uint32_t* ap = &bufA[(waveM + r * 16 + l15) * LDSR + half * 8];
#pragma unroll
            for (int i = 0; i < 4; i++) { af[r].u[i] = ap[i]; af[r].u[4 + i] = ap[4 + i]; }
        }
#pragma unroll
        for (int c = 0; c < 2; c++) {
            const uint32_t* wp = &bufB[(waveN + c * 16 + l15) * LDSR + half * 8];
#pragma unroll
            for (int v = 0; v < 8; v++) bf[c].u[v] = wp[v];
        }
#pragma unroll
        for (int r = 0; r < 2; r++)
#pragma unroll
            for (int c = 0; c < 2; c++)
                acc[r][c] = wmma_bf16(af[r].v, bf[c].v, acc[r][c]);
        __syncthreads();
        cur ^= 1;
    }
#pragma unroll
    for (int r = 0; r < 2; r++)
#pragma unroll
        for (int c = 0; c < 2; c++)
#pragma unroll
            for (int j = 0; j < 8; j++) {
                int m = rowBase0 + waveM + r * 16 + (half << 3) + j;
                int n = colBase0 + waveN + c * 16 + l15;
                float val = acc[r][c][j] + bias[n];
                if (res) val += res[(size_t)m * Nt + n];
                out[(size_t)m * Nt + n] = val;
            }
}

// ---------------------------------------------------------------------------
// Repack fused-QKV fp32 [B,N,3D] into bf16 Q[Bh,N,32], K[Bh,N,32], V^T[Bh,32,N]
// ---------------------------------------------------------------------------
__global__ void k_repack(const float* __restrict__ qkv, bf16_t* __restrict__ Qb,
                         bf16_t* __restrict__ Kb, bf16_t* __restrict__ VT) {
    int idx = blockIdx.x * 256 + threadIdx.x;          // over B*N*D
    int b = idx >> 18; int rem = idx & 0x3FFFF;
    int n = rem >> 8;  int d0 = rem & 255;
    int h = d0 >> 5;   int d = d0 & 31;
    const float* rowp = qkv + (size_t)(b * Nn + n) * 768;
    size_t qi = (((size_t)(b * Hh + h)) * Nn + n) * DH + d;
    Qb[qi] = (bf16_t)rowp[d0];
    Kb[qi] = (bf16_t)rowp[256 + d0];
    VT[(((size_t)(b * Hh + h)) * DH + d) * Nn + n] = (bf16_t)rowp[512 + d0];
}

// ---------------------------------------------------------------------------
// Flash attention with distance-decay bias.  Grid (N/128, B*H); 8 waves/block,
// each wave owns 16 query rows.  QK^T and P@V both via bf16 WMMA; online
// softmax in registers; P bounced through per-wave LDS to get A-frag layout.
// ---------------------------------------------------------------------------
__global__ __launch_bounds__(256)
void k_attn(const bf16_t* __restrict__ Qb, const bf16_t* __restrict__ Kb,
            const bf16_t* __restrict__ VT, bf16_t* __restrict__ obf) {
    __shared__ uint32_t pSm[8][16][16];   // per-wave 16x32 bf16 P tile
    int tid = threadIdx.x, lane = tid & 31, wave = tid >> 5;
    int half = lane >> 4, l15 = lane & 15;
    int bh = blockIdx.y, b = bh >> 3, h = bh & 7;
    int q0 = blockIdx.x * 128 + wave * 16;
    const float scale = 0.17677669529663687f;                 // 1/sqrt(32)
    float decay = logf(1.0f - exp2f(-2.0f - 0.5f * (float)h)); // negative

    const uint32_t* Q32 = (const uint32_t*)(Qb + (size_t)bh * Nn * DH);
    const uint32_t* K32 = (const uint32_t*)(Kb + (size_t)bh * Nn * DH);
    const uint32_t* V32 = (const uint32_t*)(VT + (size_t)bh * DH * Nn);

    BfFrag qf;
    {
        int row = q0 + l15;
        const uint32_t* qp = Q32 + row * 16 + half * 8;
#pragma unroll
        for (int i = 0; i < 4; i++) { qf.u[i] = qp[i]; qf.u[4 + i] = qp[4 + i]; }
    }
    v8f o0 = {}, o1 = {};
    float mst[8], lst[8];
#pragma unroll
    for (int j = 0; j < 8; j++) { mst[j] = -1e30f; lst[j] = 0.0f; }
    bf16_t* ps = (bf16_t*)&pSm[wave][0][0];

    for (int j0 = 0; j0 < Nn; j0 += 32) {
        BfFrag bk0, bk1;
        {
            const uint32_t* kp = K32 + (j0 + l15) * 16 + half * 8;
#pragma unroll
            for (int v = 0; v < 8; v++) bk0.u[v] = kp[v];
        }
        {
            const uint32_t* kp = K32 + (j0 + 16 + l15) * 16 + half * 8;
#pragma unroll
            for (int v = 0; v < 8; v++) bk1.u[v] = kp[v];
        }
        v8f z = {};
        v8f s0 = wmma_bf16(qf.v, bk0.v, z);
        v8f s1 = wmma_bf16(qf.v, bk1.v, z);

#pragma unroll
        for (int j = 0; j < 8; j++) {
            float iq = (float)(q0 + (half << 3) + j);
            float e0 = s0[j] * scale + fabsf(iq - (float)(j0 + l15)) * decay;
            float e1 = s1[j] * scale + fabsf(iq - (float)(j0 + 16 + l15)) * decay;
            float rm = fmaxf(e0, e1);
            for (int o = 8; o; o >>= 1) rm = fmaxf(rm, __shfl_xor(rm, o, 16));
            float mnew = fmaxf(mst[j], rm);
            float al = __expf(mst[j] - mnew);
            float p0 = __expf(e0 - mnew), p1 = __expf(e1 - mnew);
            float rs = p0 + p1;
            for (int o = 8; o; o >>= 1) rs += __shfl_xor(rs, o, 16);
            lst[j] = lst[j] * al + rs;
            mst[j] = mnew;
            o0[j] *= al; o1[j] *= al;
            int m = (half << 3) + j;
            ps[m * 32 + l15]      = (bf16_t)p0;
            ps[m * 32 + 16 + l15] = (bf16_t)p1;
        }
        __syncthreads();
        BfFrag pf;
        {
            const uint32_t* pp = &pSm[wave][l15][half * 8];
#pragma unroll
            for (int i = 0; i < 4; i++) { pf.u[i] = pp[i]; pf.u[4 + i] = pp[4 + i]; }
        }
        BfFrag bv0, bv1;
        {
            const uint32_t* vp = V32 + l15 * 512 + (j0 >> 1) + half * 8;
#pragma unroll
            for (int v = 0; v < 8; v++) bv0.u[v] = vp[v];
        }
        {
            const uint32_t* vp = V32 + (16 + l15) * 512 + (j0 >> 1) + half * 8;
#pragma unroll
            for (int v = 0; v < 8; v++) bv1.u[v] = vp[v];
        }
        o0 = wmma_bf16(pf.v, bv0.v, o0);
        o1 = wmma_bf16(pf.v, bv1.v, o1);
        __syncthreads();
    }
#pragma unroll
    for (int j = 0; j < 8; j++) {
        float inv = 1.0f / lst[j];
        int qg = q0 + (half << 3) + j;
        size_t bo = ((size_t)(b * Nn + qg)) * Dd + h * DH;
        obf[bo + l15]      = (bf16_t)(o0[j] * inv);
        obf[bo + 16 + l15] = (bf16_t)(o1[j] * inv);
    }
}

// ---------------------------------------------------------------------------
// Depthwise 3x3 conv (32x32 grid, channel-last) + bias + GELU gate -> bf16.
// One block per pixel; thread t handles gate channels {t, t+256}.
// ---------------------------------------------------------------------------
__global__ __launch_bounds__(256)
void k_dwgate(const float* __restrict__ hin, const float* __restrict__ w,
              const float* __restrict__ bc, bf16_t* __restrict__ g) {
    int pix = blockIdx.x;              // B*N pixels
    int b = pix >> 10, p = pix & 1023;
    int y = p >> 5, x = p & 31;
    int tid = threadIdx.x;
#pragma unroll
    for (int rep = 0; rep < 2; rep++) {
        int c1 = tid + rep * 256;      // gate channel 0..511
        int c2 = c1 + 512;
        float a1 = bc[c1], a2 = bc[c2];
        for (int ky = -1; ky <= 1; ky++) {
            int yy = y + ky; if ((unsigned)yy >= 32u) continue;
            for (int kx = -1; kx <= 1; kx++) {
                int xx = x + kx; if ((unsigned)xx >= 32u) continue;
                size_t idx = ((size_t)(b * Nn + (yy << 5) + xx)) * H4;
                int wi = (ky + 1) * 3 + (kx + 1);
                a1 += w[c1 * 9 + wi] * hin[idx + c1];
                a2 += w[c2 * 9 + wi] * hin[idx + c2];
            }
        }
        float ge = 0.5f * a1 * (1.0f + erff(a1 * 0.70710678f));   // exact gelu
        g[(size_t)pix * 512 + c1] = (bf16_t)(ge * a2);
    }
}

// ---------------------------------------------------------------------------
// Host orchestration (all launches on `stream`; workspace bump-allocated)
// ---------------------------------------------------------------------------
extern "C" void kernel_launch(void* const* d_in, const int* in_sizes, int n_in,
                              void* d_out, int out_size, void* d_ws, size_t ws_size,
                              hipStream_t stream) {
    (void)in_sizes; (void)n_in; (void)out_size; (void)ws_size;
    const float* x0    = (const float*)d_in[0];
    const float* qkvW  = (const float*)d_in[1];
    const float* qkvB  = (const float*)d_in[2];
    const float* projW = (const float*)d_in[3];
    const float* projB = (const float*)d_in[4];
    const float* ln1g  = (const float*)d_in[5];
    const float* ln1b  = (const float*)d_in[6];
    const float* ln2g  = (const float*)d_in[7];
    const float* ln2b  = (const float*)d_in[8];
    const float* pinW  = (const float*)d_in[9];
    const float* pinBv = (const float*)d_in[10];
    const float* dwW   = (const float*)d_in[11];
    const float* dwB   = (const float*)d_in[12];
    const float* poutW = (const float*)d_in[13];
    const float* poutBv= (const float*)d_in[14];

    char* base = (char*)d_ws; size_t off = 0;
    auto bump = [&](size_t bytes) -> char* {
        char* r = base + off; off += (bytes + 255) & ~(size_t)255; return r;
    };
    bf16_t* xn     = (bf16_t*)bump((size_t)Mm * Dd * 2);
    float*  qkvO   = (float*) bump((size_t)Mm * 768 * 4);
    bf16_t* Qb     = (bf16_t*)bump((size_t)Mm * Dd * 2);
    bf16_t* Kb     = (bf16_t*)bump((size_t)Mm * Dd * 2);
    bf16_t* VT     = (bf16_t*)bump((size_t)Mm * Dd * 2);
    bf16_t* obf    = (bf16_t*)bump((size_t)Mm * Dd * 2);
    float*  xA     = (float*) bump((size_t)Mm * Dd * 4);
    float*  xB     = (float*) bump((size_t)Mm * Dd * 4);
    float*  hbuf   = (float*) bump((size_t)Mm * H4 * 4);
    bf16_t* gbf    = (bf16_t*)bump((size_t)Mm * 512 * 2);
    bf16_t* qkvWT  = (bf16_t*)bump((size_t)2 * 768 * Dd * 2);
    bf16_t* projWT = (bf16_t*)bump((size_t)2 * Dd * Dd * 2);
    bf16_t* pinWb  = (bf16_t*)bump((size_t)2 * H4 * Dd * 2);
    bf16_t* poutWb = (bf16_t*)bump((size_t)2 * Dd * 512 * 2);

    // Weight prep (bf16 convert; [in,out] weights also transposed to [out,in])
    for (int l = 0; l < 2; l++) {
        k_transpose_bf16<<<(768 * Dd) / 256, 256, 0, stream>>>(
            qkvW + (size_t)l * Dd * 768, qkvWT + (size_t)l * 768 * Dd, Dd, 768);
        k_transpose_bf16<<<(Dd * Dd) / 256, 256, 0, stream>>>(
            projW + (size_t)l * Dd * Dd, projWT + (size_t)l * Dd * Dd, Dd, Dd);
        k_convert_bf16<<<(H4 * Dd) / 256, 256, 0, stream>>>(
            pinW + (size_t)l * H4 * Dd, pinWb + (size_t)l * H4 * Dd, H4 * Dd);
        k_convert_bf16<<<(Dd * 512) / 256, 256, 0, stream>>>(
            poutW + (size_t)l * Dd * 512, poutWb + (size_t)l * Dd * 512, Dd * 512);
    }

    for (int l = 0; l < 2; l++) {
        const float* xs = (l == 0) ? x0 : xB;
        float* xout = (l == 1) ? (float*)d_out : xB;

        k_ln<<<Mm, 256, 0, stream>>>(xs, ln1g + l * Dd, ln1b + l * Dd, xn);
        k_gemm<<<dim3(Mm / 64, 768 / 128), 256, 0, stream>>>(
            xn, qkvWT + (size_t)l * 768 * Dd, qkvB + l * 768, nullptr, qkvO, Dd, 768);
        k_repack<<<(Mm * Dd) / 256, 256, 0, stream>>>(qkvO, Qb, Kb, VT);
        k_attn<<<dim3(Nn / 128, Bb * Hh), 256, 0, stream>>>(Qb, Kb, VT, obf);
        k_gemm<<<dim3(Mm / 64, Dd / 128), 256, 0, stream>>>(
            obf, projWT + (size_t)l * Dd * Dd, projB + l * Dd, xs, xA, Dd, Dd);
        k_ln<<<Mm, 256, 0, stream>>>(xA, ln2g + l * Dd, ln2b + l * Dd, xn);
        k_gemm<<<dim3(Mm / 64, H4 / 128), 256, 0, stream>>>(
            xn, pinWb + (size_t)l * H4 * Dd, pinBv + l * H4, nullptr, hbuf, Dd, H4);
        k_dwgate<<<Mm, 256, 0, stream>>>(hbuf, dwW + (size_t)l * H4 * 9,
                                         dwB + l * H4, gbf);
        k_gemm<<<dim3(Mm / 64, Dd / 128), 256, 0, stream>>>(
            gbf, poutWb + (size_t)l * Dd * 512, poutBv + l * Dd, xA, xout, 512, Dd);
    }
}
